// AutoregressiveRNNBase_51986284150952
// MI455X (gfx1250) — compile-verified, matrix operand order checked by reading, not run on hardware
//
#include <hip/hip_runtime.h>

// ---------------- problem constants ----------------
#define B_   128
#define H_   1024
#define NS_  512
#define NBLK 64              // persistent blocks; each owns 16 h-columns, all 6 gates
#define NTHR 256             // 8 wave32s per block

typedef __attribute__((ext_vector_type(16))) __bf16 v16bf;
typedef __attribute__((ext_vector_type(8)))  float  v8f;

// float -> bf16 round-to-nearest-even
__device__ __forceinline__ unsigned short f2bf(float f) {
  unsigned u = __float_as_uint(f);
  u += 0x7fffu + ((u >> 16) & 1u);
  return (unsigned short)(u >> 16);
}

// (b, k) -> flat index in the packed WMMA 16-bit A-fragment layout
// (ISA 7.12.2: lane<16 holds K {0..7,16..23}, lane>=16 holds K {8..15,24..31})
__device__ __forceinline__ unsigned pack_idx(int b, int k) {
  int mt   = b >> 4;
  int kt   = k >> 5;
  int kr   = k & 31;
  int lc   = (kr >> 3) & 1;                  // lane-half select
  int e    = (kr & 7) + ((kr >> 4) << 3);    // element within v16bf
  int lane = (b & 15) + (lc << 4);
  return ((((unsigned)(mt * 32 + kt)) * 32u + (unsigned)lane) << 4) + (unsigned)e;
}

// ---------------- weight pack: fp32 (3H,H)x2 -> bf16 B-fragment layout ----------------
// Wpk fragment index = ((jt*32 + kt)*32 + lane)*16 + e ; jt 0..383 over [W_ih ; W_hh]
__global__ void pack_w_kernel(const float* __restrict__ Wih,
                              const float* __restrict__ Whh,
                              unsigned short* __restrict__ Wpk) {
  unsigned pidx = blockIdx.x * blockDim.x + threadIdx.x;   // exactly 6H*H threads
  int e    = pidx & 15;
  int lane = (pidx >> 4) & 31;
  int kt   = (pidx >> 9) & 31;
  int jt   = pidx >> 14;                                   // 0..383
  int j    = jt * 16 + (lane & 15);                        // gate row of Wcat (output col)
  int k    = kt * 32 + (e & 7) + ((e >> 3) << 4) + ((lane >> 4) << 3);
  float w  = (j < 3 * H_) ? Wih[(size_t)j * H_ + k]
                          : Whh[(size_t)(j - 3 * H_) * H_ + k];
  Wpk[pidx] = f2bf(w);
}

// ---------------- init: pack x0/h0 to A-fragments, write xs[:,0,:] ----------------
__global__ void init_kernel(const float* __restrict__ x0,
                            const float* __restrict__ h0,
                            float* __restrict__ out,
                            unsigned short* __restrict__ Ax,
                            unsigned short* __restrict__ Ah) {
  int tid = blockIdx.x * blockDim.x + threadIdx.x;         // B_*H_ threads
  int b = tid >> 10, j = tid & (H_ - 1);
  float xv = x0[tid];
  float hv = h0[tid];
  out[(size_t)b * NS_ * H_ + j] = xv;                      // xs[b,0,:]
  unsigned p = pack_idx(b, j);
  Ax[p] = f2bf(xv);
  Ah[p] = f2bf(hv);
}

// ---------------- device-scope barrier (atomic counter + generation spin) ----------------
__device__ __forceinline__ void grid_barrier(unsigned* cnt, volatile unsigned* gen) {
  __syncthreads();
  if (threadIdx.x == 0) {
    __threadfence();
    unsigned g = *gen;
    if (atomicAdd(cnt, 1u) == NBLK - 1) {
      *cnt = 0u;
      __threadfence();
      atomicExch((unsigned*)gen, g + 1u);
    } else {
      while (*gen == g) { __builtin_amdgcn_s_sleep(1); }
    }
    __threadfence();
  }
  __syncthreads();
}

// ---------------- persistent GRU: one grid-sync per step ----------------
// Block blk owns h-columns [blk*16, blk*16+16). It computes ALL SIX gate tiles
// for that slice (jt = blk + {0,64,128,192,256,320}), so the GRU elementwise
// fuses directly on the WMMA accumulators; h itself lives in VGPRs all 512 steps.
__global__ void __launch_bounds__(NTHR, 1)
gru_persist(const unsigned short* __restrict__ Wpk,
            unsigned short* __restrict__ Ax,
            unsigned short* __restrict__ Ah,
            const float* __restrict__ h0,
            const float* __restrict__ bih,
            const float* __restrict__ bhh,
            float* __restrict__ out,
            unsigned* __restrict__ bar) {
  extern __shared__ unsigned short sW[];   // 6 gate tiles * 32KB = 192 KB
  const int blk  = blockIdx.x;
  const int tid  = threadIdx.x;
  const int wid  = tid >> 5;               // wave owns batch rows [wid*16, wid*16+16)
  const int lane = tid & 31;

  // Stage this block's six 32KB weight tiles into LDS ONCE (reused 511 steps).
  #pragma unroll
  for (int g = 0; g < 6; ++g) {
    int jt = blk + g * 64;
    const uint4* src = (const uint4*)(Wpk + (size_t)jt * 16384u);
    uint4*       dst = (uint4*)(sW + (size_t)g * 16384u);
    for (int i = tid; i < 16384 / 8; i += NTHR) dst[i] = src[i];
  }
  __syncthreads();

  // Per-lane fixed coordinates.
  const int j       = blk * 16 + (lane & 15);              // h column
  const int rowbase = wid * 16 + ((lane >> 4) << 3);       // first of 8 batch rows
  // Combined gate biases (constant across steps).
  const float br  = bih[j]          + bhh[j];
  const float bz  = bih[H_ + j]     + bhh[H_ + j];
  const float bni = bih[2 * H_ + j];
  const float bnh = bhh[2 * H_ + j];

  // Recurrent state h in registers (C/D fragment mapping).
  float hreg[8];
  #pragma unroll
  for (int v = 0; v < 8; ++v) hreg[v] = h0[(size_t)(rowbase + v) * H_ + j];

  // Packed-store base for h_{t} -> next step's A fragments (mt == wid).
  const int kt_j = j >> 5, kr = j & 31;
  const int lc   = (kr >> 3) & 1;
  const int e_j  = (kr & 7) + ((kr >> 4) << 3);
  const unsigned p0 = (((unsigned)(wid * 32 + kt_j) * 32u
                        + (unsigned)(((lane >> 4) << 3) + (lc << 4))) << 4) + (unsigned)e_j;

  const unsigned short* Axbase = Ax + (((unsigned)(wid * 32)) * 32u + lane) * 16u;
  const unsigned short* Ahbase = Ah + (((unsigned)(wid * 32)) * 32u + lane) * 16u;
  unsigned* cnt = bar;
  volatile unsigned* gen = bar + 1;

  for (int t = 1; t < NS_; ++t) {
    // For t>=2, x == h: alias the second A stream onto the first (L0 hits).
    const unsigned short* A2base = (t == 1) ? Ahbase : Axbase;

    v8f c0 = {}, c1 = {}, c2 = {}, c3 = {}, c4 = {}, c5 = {};
    #pragma unroll 4
    for (int kt = 0; kt < 32; ++kt) {
      v16bf ax = *(const v16bf*)(Axbase + kt * 512);
      v16bf ah = *(const v16bf*)(A2base + kt * 512);
      __builtin_prefetch(Axbase + (kt + 1) * 512, 0, 3);
      const unsigned short* Bb = sW + (unsigned)kt * 512u + (unsigned)lane * 16u;
      v16bf b0 = *(const v16bf*)(Bb);
      v16bf b1 = *(const v16bf*)(Bb + 1 * 16384);
      v16bf b2 = *(const v16bf*)(Bb + 2 * 16384);
      v16bf b3 = *(const v16bf*)(Bb + 3 * 16384);
      v16bf b4 = *(const v16bf*)(Bb + 4 * 16384);
      v16bf b5 = *(const v16bf*)(Bb + 5 * 16384);
      c0 = __builtin_amdgcn_wmma_f32_16x16x32_bf16(false, ax, false, b0, (short)0, c0, false, false);
      c1 = __builtin_amdgcn_wmma_f32_16x16x32_bf16(false, ax, false, b1, (short)0, c1, false, false);
      c2 = __builtin_amdgcn_wmma_f32_16x16x32_bf16(false, ax, false, b2, (short)0, c2, false, false);
      c3 = __builtin_amdgcn_wmma_f32_16x16x32_bf16(false, ah, false, b3, (short)0, c3, false, false);
      c4 = __builtin_amdgcn_wmma_f32_16x16x32_bf16(false, ah, false, b4, (short)0, c4, false, false);
      c5 = __builtin_amdgcn_wmma_f32_16x16x32_bf16(false, ah, false, b5, (short)0, c5, false, false);
    }

    // ---- fused GRU elementwise on the accumulators (no gate scratch) ----
    #pragma unroll
    for (int v = 0; v < 8; ++v) {
      float r = 1.0f / (1.0f + __expf(-(c0[v] + c3[v] + br)));
      float z = 1.0f / (1.0f + __expf(-(c1[v] + c4[v] + bz)));
      float n = tanhf(c2[v] + bni + r * (c5[v] + bnh));
      float hnew = (1.0f - z) * n + z * hreg[v];
      hreg[v] = hnew;
      out[((size_t)(rowbase + v) * NS_ + t) * H_ + j] = hnew;   // xs[b,t,:]
      unsigned short bv = f2bf(hnew);
      Ax[p0 + (unsigned)v * 16u] = bv;                          // x_{t+1} = h_t
      Ah[p0 + (unsigned)v * 16u] = bv;                          // h_{t+1} = h_t
    }
    if (t == NS_ - 1) {
      #pragma unroll
      for (int v = 0; v < 8; ++v)
        out[(size_t)B_ * NS_ * H_ + (size_t)(rowbase + v) * H_ + j] = hreg[v];
    }

    grid_barrier(cnt, gen);   // single sync: publish h_t before next step's GEMM
  }
}

// ---------------- launch ----------------
extern "C" void kernel_launch(void* const* d_in, const int* in_sizes, int n_in,
                              void* d_out, int out_size, void* d_ws, size_t ws_size,
                              hipStream_t stream) {
  (void)in_sizes; (void)n_in; (void)out_size; (void)ws_size;
  const float* x0  = (const float*)d_in[0];
  const float* h0  = (const float*)d_in[1];
  const float* Wih = (const float*)d_in[3];
  const float* Whh = (const float*)d_in[4];
  const float* bih = (const float*)d_in[5];
  const float* bhh = (const float*)d_in[6];
  float* out = (float*)d_out;

  char* ws = (char*)d_ws;
  unsigned*       bar = (unsigned*)ws;                                   // 256 B
  unsigned short* Wpk = (unsigned short*)(ws + 256);                     // 12 MB bf16
  unsigned short* Ax  = (unsigned short*)(ws + 256 + 12582912);          // 256 KB
  unsigned short* Ah  = (unsigned short*)(ws + 256 + 12582912 + 262144); // 256 KB

  hipMemsetAsync(d_ws, 0, 256, stream);  // barrier count + generation = 0

  pack_w_kernel<<<(6 * H_ * H_) / NTHR, NTHR, 0, stream>>>(Wih, Whh, Wpk);
  init_kernel<<<(B_ * H_) / NTHR, NTHR, 0, stream>>>(x0, h0, out, Ax, Ah);
  gru_persist<<<NBLK, NTHR, 196608 /* 192 KB dynamic LDS */, stream>>>(
      Wpk, Ax, Ah, h0, bih, bhh, out, bar);
}
